// OpenLSTML4casadi_65051574665297
// MI455X (gfx1250) — matrix operand order repaired and checked by reading, not compiled, not run on hardware
//
#include <hip/hip_runtime.h>
#include <hip/hip_bf16.h>
#include <math.h>

#define SEQ   131072
#define NCTX  65536
#define NIN   64
#define H     128
#define NG    512      // 4*H gate width
#define USTRIDE 65     // u_train row: 64 inputs + 1 observed y

typedef __attribute__((ext_vector_type(16))) __bf16 v16bf;
typedef __attribute__((ext_vector_type(8)))  float  v8f;

__device__ __forceinline__ float sigm(float x) {
    return 1.0f / (1.0f + __expf(-x));
}

// ---------------------------------------------------------------------------
// Kernel 1: G[t][n] = sum_k u[t][k] * W_ih[n][k] + (b_ih[n] + b_hh[n])
// One wave per 16-row M-tile; 32 N-tiles of 16; K=64 via two bf16 WMMAs.
// W_ih staged in LDS as bf16 (512*64*2 = 64KB).
// ---------------------------------------------------------------------------
__global__ __launch_bounds__(256) void gates_gemm_wmma(
    const float* __restrict__ u,
    const float* __restrict__ Wih,
    const float* __restrict__ bih,
    const float* __restrict__ bhh,
    float* __restrict__ G)
{
    __shared__ __bf16 Wlds[NG * NIN];   // W[n][k] at n*64+k, exactly 64 KB

    const int tid = threadIdx.x;
    for (int idx = tid; idx < NG * NIN; idx += 256)
        Wlds[idx] = (__bf16)Wih[idx];
    __syncthreads();

    const int wave = tid >> 5;
    const int lane = tid & 31;
    const int half = lane >> 4;      // 0: lanes 0-15, 1: lanes 16-31
    const int lrow = lane & 15;
    const int m0   = (blockIdx.x * 8 + wave) * 16;
    const int mrow = m0 + lrow;
    const float* urow = u + (size_t)mrow * USTRIDE;

    // A fragment (16x32 bf16): lanes 0-15 row M=lrow hold
    // K = base+[0..15] in elems 0..15;
    // lanes 16-31 hold base+[16..31] (contiguous, matches B).
    v16bf a0, a1;
    {
        const int k1a = half * 16;
        const int k2a = half * 16 + 8;
#pragma unroll
        for (int i = 0; i < 8; ++i) {
            a0[i]     = (__bf16)urow[k1a + i];
            a0[8 + i] = (__bf16)urow[k2a + i];
            a1[i]     = (__bf16)urow[32 + k1a + i];
            a1[8 + i] = (__bf16)urow[32 + k2a + i];
        }
    }

    for (int nt = 0; nt < 32; ++nt) {
        const int n = nt * 16 + lrow;
        const float bias = bih[n] + bhh[n];   // same for every row of the tile

        // B fragment (32x16): lane column N=lrow; lanes 0-15 hold K=kb*32+[0..15],
        // lanes 16-31 hold K=kb*32+[16..31]; contiguous in Wlds row n.
        const v16bf b0 = *(const v16bf*)&Wlds[n * NIN +      half * 16];
        const v16bf b1 = *(const v16bf*)&Wlds[n * NIN + 32 + half * 16];

        v8f acc;
#pragma unroll
        for (int r = 0; r < 8; ++r) acc[r] = bias;

        acc = __builtin_amdgcn_wmma_f32_16x16x32_bf16(false, a0, false, b0,
                                                      (short)0, acc, false, false);
        acc = __builtin_amdgcn_wmma_f32_16x16x32_bf16(false, a1, false, b1,
                                                      (short)0, acc, false, false);

        // D layout: lane holds (M = r + 8*half, N = lrow), r = VGPR index.
        float* grow = G + (size_t)(m0 + half * 8) * NG + n;
#pragma unroll
        for (int r = 0; r < 8; ++r)
            grow[(size_t)r * NG] = acc[r];
    }
}

// ---------------------------------------------------------------------------
// Kernel 2: teacher-forced phase. h_prev == observed y_t, so the 128 cell
// channels are fully independent serial recurrences: one channel per thread,
// zero synchronization. Emits P[t][j] = sigm(o)*tanh(c)*W_hr[j] and c_final.
// ---------------------------------------------------------------------------
__global__ __launch_bounds__(128) void est_cell_scan(
    const float* __restrict__ u,
    const float* __restrict__ G,
    const float* __restrict__ Whh,
    const float* __restrict__ Whr,
    float* __restrict__ P,
    float* __restrict__ Cfin)
{
    const int j = threadIdx.x;          // channel 0..127
    const float wi = Whh[j];
    const float wf = Whh[j + 128];
    const float wg = Whh[j + 256];
    const float wo = Whh[j + 384];
    const float wr = Whr[j];

    float c = 0.0f;
    for (int t = 0; t < NCTX; ++t) {
        const float* gt = G + (size_t)t * NG;
        __builtin_prefetch(gt + (size_t)24 * NG, 0, 1);
        const float yv = u[(size_t)t * USTRIDE + NIN];   // observed y_t
        const float ig = sigm(gt[j]       + wi * yv);
        const float fg = sigm(gt[j + 128] + wf * yv);
        const float gg = tanhf(gt[j + 256] + wg * yv);
        const float og = sigm(gt[j + 384] + wo * yv);
        c = fg * c + ig * gg;
        P[(size_t)t * H + j] = og * tanhf(c) * wr;
    }
    Cfin[j] = c;
}

// ---------------------------------------------------------------------------
// Kernel 3: y1[t] = sum_j P[t][j], fully parallel. One wave per timestep.
// ---------------------------------------------------------------------------
__global__ __launch_bounds__(256) void est_output_reduce(
    const float* __restrict__ P,
    float* __restrict__ out)
{
    const int wave = threadIdx.x >> 5;
    const int lane = threadIdx.x & 31;
    const int t    = blockIdx.x * 8 + wave;

    const float* pt = P + (size_t)t * H;
    float s = pt[lane] + pt[lane + 32] + pt[lane + 64] + pt[lane + 96];
#pragma unroll
    for (int off = 16; off > 0; off >>= 1)
        s += __shfl_xor(s, off, 32);
    if (lane == 0) out[t] = s;
}

// ---------------------------------------------------------------------------
// Kernel 4: prediction phase — irreducibly sequential (h feeds back through
// all channels). Single wave32, 4 channels/lane, shuffle reduction per step.
// ---------------------------------------------------------------------------
__global__ __launch_bounds__(32) void pred_scan(
    const float* __restrict__ G,
    const float* __restrict__ Whh,
    const float* __restrict__ Whr,
    const float* __restrict__ P,
    const float* __restrict__ Cfin,
    float* __restrict__ out)
{
    const int lane = threadIdx.x;

    float wi[4], wf[4], wg[4], wo[4], wr[4], cc[4];
#pragma unroll
    for (int s = 0; s < 4; ++s) {
        const int j = lane + 32 * s;
        wi[s] = Whh[j];
        wf[s] = Whh[j + 128];
        wg[s] = Whh[j + 256];
        wo[s] = Whh[j + 384];
        wr[s] = Whr[j];
        cc[s] = Cfin[j];
    }

    // h_n = y1[NCTX-1] = sum_j P[NCTX-1][j]
    const float* plast = P + (size_t)(NCTX - 1) * H;
    float h = plast[lane] + plast[lane + 32] + plast[lane + 64] + plast[lane + 96];
#pragma unroll
    for (int off = 16; off > 0; off >>= 1)
        h += __shfl_xor(h, off, 32);

    for (int t = NCTX; t < SEQ; ++t) {
        const float* gt = G + (size_t)t * NG;
        __builtin_prefetch(gt + (size_t)16 * NG, 0, 1);
        float term = 0.0f;
#pragma unroll
        for (int s = 0; s < 4; ++s) {
            const int j = lane + 32 * s;
            const float ig = sigm(gt[j]       + wi[s] * h);
            const float fg = sigm(gt[j + 128] + wf[s] * h);
            const float gg = tanhf(gt[j + 256] + wg[s] * h);
            const float og = sigm(gt[j + 384] + wo[s] * h);
            cc[s] = fg * cc[s] + ig * gg;
            term += og * tanhf(cc[s]) * wr[s];
        }
#pragma unroll
        for (int off = 16; off > 0; off >>= 1)
            term += __shfl_xor(term, off, 32);
        h = term;                        // reduction leaves sum in all lanes
        if (lane == 0) out[t] = h;
    }
}

extern "C" void kernel_launch(void* const* d_in, const int* in_sizes, int n_in,
                              void* d_out, int out_size, void* d_ws, size_t ws_size,
                              hipStream_t stream) {
    const float* u   = (const float*)d_in[0];   // (SEQ, 1, 65)
    const float* Wih = (const float*)d_in[1];   // (512, 64)
    const float* Whh = (const float*)d_in[2];   // (512, 1)
    const float* bih = (const float*)d_in[3];   // (512,)
    const float* bhh = (const float*)d_in[4];   // (512,)
    const float* Whr = (const float*)d_in[5];   // (1, 128)
    float* out = (float*)d_out;                 // (SEQ,)

    // Workspace layout
    float* G    = (float*)d_ws;                         // SEQ*512 f32 = 256 MB
    float* P    = G + (size_t)SEQ * NG;                 // NCTX*128 f32 = 32 MB
    float* Cfin = P + (size_t)NCTX * H;                 // 128 f32

    gates_gemm_wmma<<<dim3(SEQ / 16 / 8), 256, 0, stream>>>(u, Wih, bih, bhh, G);
    est_cell_scan<<<dim3(1), 128, 0, stream>>>(u, G, Whh, Whr, P, Cfin);
    est_output_reduce<<<dim3(NCTX / 8), 256, 0, stream>>>(P, out);
    pred_scan<<<dim3(1), 32, 0, stream>>>(G, Whh, Whr, P, Cfin, out);
}
